// Encoder_65051574665837
// MI455X (gfx1250) — matrix operand order repaired and checked by reading, not compiled, not run on hardware
//
#include <hip/hip_runtime.h>
#include <hip/hip_bf16.h>

#define VOCAB 32000
#define E_DIM 512
#define H_DIM 1024
#define B_DIM 32
#define S_DIM 512
#define G_DIM (3 * H_DIM)          // 3072
#define M_DIM (B_DIM * S_DIM)      // 16384
#define NWG_SCAN 32                // one WG per 32-col H slice

typedef __attribute__((ext_vector_type(16))) __bf16 v16bf;
typedef __attribute__((ext_vector_type(16))) unsigned short v16u;
typedef __attribute__((ext_vector_type(8)))  float v8f;
typedef __attribute__((ext_vector_type(4)))  unsigned int v4u;
typedef __attribute__((ext_vector_type(8)))  int v8i;
typedef __attribute__((ext_vector_type(4)))  int v4i;

// ---------- helpers ----------
__device__ __forceinline__ unsigned short f2bf_rne(float f) {
  union { float f; unsigned u; } c; c.f = f;
  unsigned u = c.u;
  unsigned r = u + 0x7FFFu + ((u >> 16) & 1u);   // round-to-nearest-even
  return (unsigned short)(r >> 16);
}
__device__ __forceinline__ float bf2f(unsigned short h) {
  union { unsigned u; float f; } c; c.u = ((unsigned)h) << 16;
  return c.f;
}
__device__ __forceinline__ v8f wmma_bf16(v16bf a, v16bf b, v8f c) {
  return __builtin_amdgcn_wmma_f32_16x16x32_bf16(false, a, false, b, (short)0, c,
                                                 false, false);
}

// TDM: 2D tile, data_size=8B, 4 rows of 1024 elems (8KB), row stride 4096 elems
// (32KB). Loads one 32KB chunk of the h-state fragment buffer into LDS.
__device__ __forceinline__ void tdm_load_chunk(unsigned long long gaddr,
                                               unsigned lds_off) {
  v4u g0;
  g0[0] = 1u;                                           // count=1, user desc
  g0[1] = lds_off;                                      // lds_addr
  g0[2] = (unsigned)(gaddr & 0xFFFFFFFFull);            // global_addr[31:0]
  g0[3] = (unsigned)((gaddr >> 32) & 0x01FFFFFFull)     // global_addr[56:32]
          | 0x80000000u;                                // type=2 ("image")
  v8i g1;
  g1[0] = (int)(3u << 16);        // workgroup_mask=0, data_size=3 (8B)
  g1[1] = (int)(4096u << 16);     // tensor_dim0 = 4096 (bits 79:48)
  g1[2] = (int)(4u << 16);        // tensor_dim1 = 4    (bits 111:80)
  g1[3] = (int)(1024u << 16);     // tile_dim0 = 1024   (bits 127:112)
  g1[4] = 4;                      // tile_dim1 = 4      (bits 143:128)
  g1[5] = 4096;                   // tensor_dim0_stride = 4096 (bits 207:160)
  g1[6] = 0;
  g1[7] = 0;
  v4i gz4 = {0, 0, 0, 0};
  v8i gz8 = {0, 0, 0, 0, 0, 0, 0, 0};
  __builtin_amdgcn_tensor_load_to_lds(g0, g1, gz4, gz4, gz8, 0);
}

// ---------- K1: pack weight matrix W[G rows][K cols] (row-major fp32) into
// B-fragment layout, split hi/lo bf16.
// B-tile (nt, kt): lane l holds col n = l&15, K = kt*32 + (e<8?0:16) + (l<16?0:8) + (e&7)
// packed index: ((nt*KT + kt)*32 + lane)*16 + e
__global__ void k_pack_w(const float* __restrict__ W, int kdim, int ktiles,
                         unsigned short* __restrict__ ohi,
                         unsigned short* __restrict__ olo, int ntiles_total) {
  int gid  = blockIdx.x * blockDim.x + threadIdx.x;
  int tile = gid >> 5;
  if (tile >= ntiles_total) return;
  int lane = gid & 31;
  int nt = tile / ktiles;
  int kt = tile % ktiles;
  int row   = nt * 16 + (lane & 15);
  int kbase = kt * 32 + ((lane & 16) ? 8 : 0);
  const float* src = W + (size_t)row * kdim + kbase;
  v16u hi, lo;
#pragma unroll
  for (int e = 0; e < 16; ++e) {
    float f = src[((e < 8) ? 0 : 16) + (e & 7)];
    unsigned short h = f2bf_rne(f);
    hi[e] = h;
    lo[e] = f2bf_rne(f - bf2f(h));
  }
  size_t off = ((size_t)tile * 32 + lane) * 16;
  *(v16u*)(ohi + off) = hi;
  *(v16u*)(olo + off) = lo;
}

// ---------- K2: embedding gather + pack into A-fragment layout (split bf16).
// A-tile (mt, kt): lane l holds row m = mt*16 + (l&15);
// element e -> K = kt*32 + (e<8?0:16) + (l<16?0:8) + (e&7)
__global__ void k_embed_pack(const int* __restrict__ x, const float* __restrict__ emb,
                             unsigned short* __restrict__ ahi,
                             unsigned short* __restrict__ alo) {
  int gid  = blockIdx.x * blockDim.x + threadIdx.x;
  int tile = gid >> 5;
  if (tile >= (M_DIM / 16) * (E_DIM / 32)) return;
  int lane = gid & 31;
  int mt = tile >> 4;
  int kt = tile & 15;
  int m   = mt * 16 + (lane & 15);
  int tok = x[m];
  const float* row = emb + (size_t)tok * E_DIM + kt * 32;
  int base0 = (lane & 16) ? 8 : 0;
  v16u hi, lo;
#pragma unroll
  for (int e = 0; e < 16; ++e) {
    int k = ((e < 8) ? 0 : 16) + base0 + (e & 7);
    float f = row[k];
    unsigned short h = f2bf_rne(f);
    hi[e] = h;
    lo[e] = f2bf_rne(f - bf2f(h));
  }
  size_t off = ((size_t)tile * 32 + lane) * 16;
  *(v16u*)(ahi + off) = hi;
  *(v16u*)(alo + off) = lo;
}

// ---------- K3: input GEMM xg[m,g] = xemb[m,:] . W_ih[g,:] + b_ih[g]
// stored time-major: xg[(s*B + b)*G + g], m = b*S + s.
__global__ void __launch_bounds__(256) k_gemm_in(
    const unsigned short* __restrict__ ahi, const unsigned short* __restrict__ alo,
    const unsigned short* __restrict__ bhi, const unsigned short* __restrict__ blo,
    const float* __restrict__ b_ih, float* __restrict__ xg) {
  int lane = threadIdx.x & 31;
  int wv   = threadIdx.x >> 5;
  int mt   = blockIdx.x & 1023;
  int ns   = blockIdx.x >> 10;       // 0..5
  int ntb  = ns * 32 + wv * 4;       // base N-tile (of 192)

  v8f acc[4] = {};
  for (int kt = 0; kt < 16; ++kt) {
    size_t aoff = (((size_t)mt * 16 + kt) * 32 + lane) * 16;
    v16bf a_hi = *(const v16bf*)(ahi + aoff);
    v16bf a_lo = *(const v16bf*)(alo + aoff);
#pragma unroll
    for (int j = 0; j < 4; ++j) {
      size_t boff = (((size_t)(ntb + j) * 16 + kt) * 32 + lane) * 16;
      v16bf w_hi = *(const v16bf*)(bhi + boff);
      v16bf w_lo = *(const v16bf*)(blo + boff);
      acc[j] = wmma_bf16(a_hi, w_hi, acc[j]);
      acc[j] = wmma_bf16(a_hi, w_lo, acc[j]);
      acc[j] = wmma_bf16(a_lo, w_hi, acc[j]);
    }
  }
  int mbase = mt * 16 + ((lane & 16) ? 8 : 0);
  int nl = lane & 15;
#pragma unroll
  for (int j = 0; j < 4; ++j) {
    int g = (ntb + j) * 16 + nl;
    float bias = b_ih[g];
#pragma unroll
    for (int r = 0; r < 8; ++r) {
      int m = mbase + r;
      int b = m >> 9, s = m & 511;
      xg[(size_t)(s * B_DIM + b) * G_DIM + g] = acc[j][r] + bias;
    }
  }
}

// ---------- K4: persistent GRU scan. 32 WGs x 192 threads (6 waves).
// WG w owns H cols [w*32, w*32+32) -> 6 gate N-tiles; wave v owns N-tile v for
// BOTH M-tiles, so each W_hh fragment is loaded exactly once per WG per step.
// h-state fragments (128KB) are staged into LDS by the Tensor Data Mover in
// four 32KB chunks, double-buffered, overlapped with WMMA via TENSORcnt.
__global__ void __launch_bounds__(192) k_scan(
    const unsigned short* __restrict__ whh_hi, const unsigned short* __restrict__ whh_lo,
    const float* __restrict__ xg, const float* __restrict__ b_hh,
    unsigned short* __restrict__ hfrag,   // [2 bufs][hi|lo][2*32*32*16 u16] = 2x128KB
    int* __restrict__ cnt,                // [S] arrival counters (pre-zeroed)
    float* __restrict__ out, float* __restrict__ state) {
  __shared__ unsigned short stage[2][16384];  // 2 x 32KB TDM staging buffers
  __shared__ float hg[32][96];                // r|z|n gate pre-activations
  __shared__ float hlds[32][32];              // this WG's h slice (fp32)
  const int tid  = threadIdx.x;
  const int lane = tid & 31;
  const int wv   = tid >> 5;                  // 0..5
  const int w    = blockIdx.x;                // 0..31

  for (int i = tid; i < 32 * 32; i += 192) hlds[i >> 5][i & 31] = 0.0f;
  __syncthreads();

  const int ntg  = (wv >> 1) * 64 + w * 2 + (wv & 1);  // global N-tile of 192
  const unsigned stage_base = (unsigned)(size_t)(void*)&stage[0][0];

  for (int t = 0; t < S_DIM; ++t) {
    const int rb = t & 1;
    const unsigned long long hbase =
        (unsigned long long)(size_t)hfrag + (unsigned long long)rb * 131072ull;

    v8f acc[2] = {};
    if (wv == 0) tdm_load_chunk(hbase, stage_base);           // chunk 0
    for (int c = 0; c < 4; ++c) {
      if (wv == 0) {
        if (c < 3) {
          tdm_load_chunk(hbase + (unsigned long long)(c + 1) * 8192ull,
                         stage_base + (unsigned)(((c + 1) & 1) * 32768));
          __builtin_amdgcn_s_wait_tensorcnt(1);
        } else {
          __builtin_amdgcn_s_wait_tensorcnt(0);
        }
      }
      __syncthreads();
      const unsigned short* sb = &stage[c & 1][0];
      for (int ktl = 0; ktl < 8; ++ktl) {
        int kt = c * 8 + ktl;
        // LDS rows: 0=mt0 hi, 1=mt1 hi, 2=mt0 lo, 3=mt1 lo (4096 u16 each)
        v16bf a0h = *(const v16bf*)(sb + 0 * 4096 + ktl * 512 + lane * 16);
        v16bf a1h = *(const v16bf*)(sb + 1 * 4096 + ktl * 512 + lane * 16);
        v16bf a0l = *(const v16bf*)(sb + 2 * 4096 + ktl * 512 + lane * 16);
        v16bf a1l = *(const v16bf*)(sb + 3 * 4096 + ktl * 512 + lane * 16);
        size_t boff = (((size_t)ntg * 32 + kt) * 32 + lane) * 16;
        v16bf bh = *(const v16bf*)(whh_hi + boff);
        v16bf bl = *(const v16bf*)(whh_lo + boff);
        acc[0] = wmma_bf16(a0h, bh, acc[0]);
        acc[0] = wmma_bf16(a0h, bl, acc[0]);
        acc[0] = wmma_bf16(a0l, bh, acc[0]);
        acc[1] = wmma_bf16(a1h, bh, acc[1]);
        acc[1] = wmma_bf16(a1h, bl, acc[1]);
        acc[1] = wmma_bf16(a1l, bh, acc[1]);
      }
      __syncthreads();
    }
    {
      int ml = (lane & 16) ? 8 : 0;
      int nl = lane & 15;
      int gcol = (wv >> 1) * 1024 + w * 32 + (wv & 1) * 16 + nl;
      float bias = b_hh[gcol];
#pragma unroll
      for (int r = 0; r < 8; ++r) {
        hg[ml + r][wv * 16 + nl]      = acc[0][r] + bias;  // M-tile 0
        hg[16 + ml + r][wv * 16 + nl] = acc[1][r] + bias;  // M-tile 1
      }
    }
    __syncthreads();

    // gate math: 32 rows x 32 cols = 1024 elems over 192 threads
    for (int idx = tid; idx < 1024; idx += 192) {
      int m = idx >> 5, j = idx & 31;
      size_t xb = ((size_t)t * B_DIM + m) * G_DIM + w * 32 + j;
      float xr = xg[xb], xz = xg[xb + 1024], xn = xg[xb + 2048];
      float hr = hg[m][j], hz = hg[m][32 + j], hn = hg[m][64 + j];
      float r = 1.0f / (1.0f + expf(-(xr + hr)));
      float z = 1.0f / (1.0f + expf(-(xz + hz)));
      float n = tanhf(xn + r * hn);
      float hv = (1.0f - z) * n + z * hlds[m][j];
      hlds[m][j] = hv;
      out[((size_t)m * S_DIM + t) * H_DIM + w * 32 + j] = hv;
      if (t == S_DIM - 1) state[(size_t)m * H_DIM + w * 32 + j] = hv;
    }
    __syncthreads();

    // repack h_new slice (32 cols = K-tile w) into next step's fragment buffer
    const int wb = rb ^ 1;
    unsigned short* hout_hi = hfrag + (size_t)wb * 65536;
    unsigned short* hout_lo = hout_hi + 32768;
    if (tid < 64) {
      int l   = tid & 31;
      int pmt = tid >> 5;            // M-tile 0/1
      int pm  = pmt * 16 + (l & 15);
      int base0 = (l & 16) ? 8 : 0;
      v16u hiv, lov;
#pragma unroll
      for (int e = 0; e < 16; ++e) {
        int k = ((e < 8) ? 0 : 16) + base0 + (e & 7);
        float f = hlds[pm][k];
        unsigned short h = f2bf_rne(f);
        hiv[e] = h;
        lov[e] = f2bf_rne(f - bf2f(h));
      }
      size_t off = (((size_t)pmt * 32 + w) * 32 + l) * 16;
      *(v16u*)(hout_hi + off) = hiv;
      *(v16u*)(hout_lo + off) = lov;
    }
    __syncthreads();
    __threadfence();
    if (tid == 0) {
      __hip_atomic_fetch_add(&cnt[t], 1, __ATOMIC_RELEASE, __HIP_MEMORY_SCOPE_AGENT);
      while (__hip_atomic_load(&cnt[t], __ATOMIC_ACQUIRE, __HIP_MEMORY_SCOPE_AGENT) <
             NWG_SCAN)
        __builtin_amdgcn_s_sleep(1);
    }
    __syncthreads();
  }
}

// ---------- K5: LayerNorm over last dim (1024), one wave per row, in place.
__global__ void __launch_bounds__(256) k_lnorm(float* __restrict__ out,
                                               const float* __restrict__ gamma,
                                               const float* __restrict__ beta) {
  int lane = threadIdx.x & 31;
  int wv   = threadIdx.x >> 5;
  int row  = blockIdx.x * 8 + wv;
  if (row >= M_DIM) return;
  float4* rp = (float4*)(out + (size_t)row * H_DIM);
  float4 v[8];
  float s = 0.f, q = 0.f;
#pragma unroll
  for (int i = 0; i < 8; ++i) {
    v[i] = rp[i * 32 + lane];
    s += v[i].x + v[i].y + v[i].z + v[i].w;
    q += v[i].x * v[i].x + v[i].y * v[i].y + v[i].z * v[i].z + v[i].w * v[i].w;
  }
#pragma unroll
  for (int off = 16; off; off >>= 1) {
    s += __shfl_xor(s, off, 32);
    q += __shfl_xor(q, off, 32);
  }
  float mu  = s * (1.0f / H_DIM);
  float var = q * (1.0f / H_DIM) - mu * mu;
  float rs  = rsqrtf(var + 1e-5f);
  const float4* gp = (const float4*)gamma;
  const float4* bp = (const float4*)beta;
#pragma unroll
  for (int i = 0; i < 8; ++i) {
    float4 g = gp[i * 32 + lane], b = bp[i * 32 + lane], o;
    o.x = (v[i].x - mu) * rs * g.x + b.x;
    o.y = (v[i].y - mu) * rs * g.y + b.y;
    o.z = (v[i].z - mu) * rs * g.z + b.z;
    o.w = (v[i].w - mu) * rs * g.w + b.w;
    rp[i * 32 + lane] = o;
  }
}

extern "C" void kernel_launch(void* const* d_in, const int* in_sizes, int n_in,
                              void* d_out, int out_size, void* d_ws, size_t ws_size,
                              hipStream_t stream) {
  const int*   x     = (const int*)d_in[0];
  const float* emb   = (const float*)d_in[1];
  const float* Wih   = (const float*)d_in[2];
  const float* Whh   = (const float*)d_in[3];
  const float* bih   = (const float*)d_in[4];
  const float* bhh   = (const float*)d_in[5];
  const float* gamma = (const float*)d_in[6];
  const float* beta  = (const float*)d_in[7];
  float* out   = (float*)d_out;
  float* state = out + (size_t)B_DIM * S_DIM * H_DIM;

  char* ws = (char*)d_ws;
  size_t off = 0;
  float* xg = (float*)(ws + off);                  off += (size_t)M_DIM * G_DIM * 4;
  unsigned short* wih_hi = (unsigned short*)(ws + off); off += (size_t)G_DIM * E_DIM * 2;
  unsigned short* wih_lo = (unsigned short*)(ws + off); off += (size_t)G_DIM * E_DIM * 2;
  unsigned short* whh_hi = (unsigned short*)(ws + off); off += (size_t)G_DIM * H_DIM * 2;
  unsigned short* whh_lo = (unsigned short*)(ws + off); off += (size_t)G_DIM * H_DIM * 2;
  unsigned short* a_hi   = (unsigned short*)(ws + off); off += (size_t)M_DIM * E_DIM * 2;
  unsigned short* a_lo   = (unsigned short*)(ws + off); off += (size_t)M_DIM * E_DIM * 2;
  unsigned short* hfrag  = (unsigned short*)(ws + off); off += (size_t)2 * 2 * 32768 * 2;
  int* cnt = (int*)(ws + off);                     off += (size_t)S_DIM * 4;

  // zero h_{-1} fragments + step counters (contiguous region)
  (void)hipMemsetAsync(hfrag, 0, (size_t)2 * 2 * 32768 * 2 + (size_t)S_DIM * 4, stream);

  k_pack_w<<<384, 256, 0, stream>>>(Wih, E_DIM, E_DIM / 32, wih_hi, wih_lo,
                                    (G_DIM / 16) * (E_DIM / 32));
  k_pack_w<<<768, 256, 0, stream>>>(Whh, H_DIM, H_DIM / 32, whh_hi, whh_lo,
                                    (G_DIM / 16) * (H_DIM / 32));
  k_embed_pack<<<2048, 256, 0, stream>>>(x, emb, a_hi, a_lo);
  k_gemm_in<<<6144, 256, 0, stream>>>(a_hi, a_lo, wih_hi, wih_lo, bih, xg);
  k_scan<<<NWG_SCAN, 192, 0, stream>>>(whh_hi, whh_lo, xg, bhh, hfrag, cnt, out, state);
  k_lnorm<<<2048, 256, 0, stream>>>(out, gamma, beta);
}
